// EncoderBlock_30760555774189
// MI455X (gfx1250) — compile-verified, matrix operand order checked by reading, not run on hardware
//
#include <hip/hip_runtime.h>
#include <math.h>

typedef __bf16 bf16;
typedef __attribute__((ext_vector_type(16))) __bf16 v16bf;
typedef __attribute__((ext_vector_type(8)))  __bf16 v8bf;
typedef __attribute__((ext_vector_type(8)))  float  v8f;

#define D_MODEL 1024
#define NHEAD   16
#define DKH     64
#define D_FC    4096
#define BATCH   2
#define SEQ     1024
#define M_TOK   (BATCH*SEQ)   // 2048
#define LN_EPS  1e-6f

// ---------------------------------------------------------------------------
// fragment load: 16-bit A/B fragment from a padded row in LDS/global.
// lane L: row chosen by caller (L&15), kb = (L>>4)*8.
// elements 0..7  = K kb..kb+7 ; elements 8..15 = K 16+kb..16+kb+7
// ---------------------------------------------------------------------------
__device__ __forceinline__ v16bf load_frag(const bf16* rowbase, int kb) {
    v8bf lo = *(const v8bf*)(rowbase + kb);
    v8bf hi = *(const v8bf*)(rowbase + kb + 16);
    v16bf f;
#pragma unroll
    for (int i = 0; i < 8; ++i) { f[i] = lo[i]; f[i + 8] = hi[i]; }
    return f;
}

#define WMMA_BF16(A_, B_, C_) \
    __builtin_amdgcn_wmma_f32_16x16x32_bf16(false, (A_), false, (B_), (short)0, (C_), false, false)

// CDNA5 async global->LDS copy, 16B per lane, tracked by ASYNCcnt.
// vdst = wave-relative LDS byte address, addr = per-lane 64-bit global address.
__device__ __forceinline__ void async_ld_b128(uint32_t lds_addr, const bf16* gaddr) {
    asm volatile("global_load_async_to_lds_b128 %0, %1, off"
                 :: "v"(lds_addr), "v"(gaddr) : "memory");
}
__device__ __forceinline__ uint32_t lds_off32(const void* p) {
    return (uint32_t)(uintptr_t)p;   // flat LDS aperture: low 32 bits = LDS address
}

// ---------------------------------------------------------------------------
// float -> bf16 elementwise (weights). n multiple of 4.
// ---------------------------------------------------------------------------
__global__ void cvt_f32_bf16(const float* __restrict__ in, bf16* __restrict__ out, int n4) {
    int i = blockIdx.x * blockDim.x + threadIdx.x;
    if (i >= n4) return;
    float4 v = *(const float4*)(in + (size_t)i * 4);
    bf16* o = out + (size_t)i * 4;
    o[0] = (bf16)v.x; o[1] = (bf16)v.y; o[2] = (bf16)v.z; o[3] = (bf16)v.w;
}

// ---------------------------------------------------------------------------
// LayerNorm (ddof=1, divide by std+eps), D_MODEL=1024, 256 threads/row.
// ---------------------------------------------------------------------------
__global__ __launch_bounds__(256) void layernorm_bf16(
    const float* __restrict__ x, const float* __restrict__ alpha,
    const float* __restrict__ beta, bf16* __restrict__ out)
{
    __shared__ float red[256];
    const int row = blockIdx.x;
    const int tid = threadIdx.x;
    const float* xr = x + (size_t)row * D_MODEL;

    float v[4];
    float s = 0.f;
#pragma unroll
    for (int i = 0; i < 4; ++i) { v[i] = xr[tid + i * 256]; s += v[i]; }
    red[tid] = s; __syncthreads();
#pragma unroll
    for (int off = 128; off > 0; off >>= 1) {
        if (tid < off) red[tid] += red[tid + off];
        __syncthreads();
    }
    const float mean = red[0] * (1.0f / 1024.0f);
    __syncthreads();

    float s2 = 0.f;
#pragma unroll
    for (int i = 0; i < 4; ++i) { float d = v[i] - mean; s2 += d * d; }
    red[tid] = s2; __syncthreads();
#pragma unroll
    for (int off = 128; off > 0; off >>= 1) {
        if (tid < off) red[tid] += red[tid + off];
        __syncthreads();
    }
    const float var = red[0] * (1.0f / 1023.0f);   // ddof = 1
    const float inv = 1.0f / (sqrtf(var) + LN_EPS);

    bf16* orow = out + (size_t)row * D_MODEL;
#pragma unroll
    for (int i = 0; i < 4; ++i) {
        int c = tid + i * 256;
        orow[c] = (bf16)(alpha[c] * (v[i] - mean) * inv + beta[c]);
    }
}

// ---------------------------------------------------------------------------
// Tiled WMMA bf16 GEMM:  C[M,N] = A[M,K] * B[N,K]^T   (+bias/relu/res)
// block tile 128x128, 8 waves (4 along M x 2 along N), wave tile 32x64.
// Double-buffered LDS staged via async global->LDS copies (ASYNCcnt pipeline).
// ---------------------------------------------------------------------------
template<int HAS_BIAS, int HAS_RELU, int HAS_RES, int OUT_BF16>
__global__ __launch_bounds__(256) void gemm_bf16_wmma(
    const bf16* __restrict__ A, const bf16* __restrict__ B,
    const float* __restrict__ bias, const float* __restrict__ res,
    float* __restrict__ Cf, bf16* __restrict__ Cb,
    int M, int N, int K)
{
    const int LDT = 40;                     // padded row stride (halfs): 80B = 5*16B
    __shared__ __align__(16) bf16 As[2][128 * 40];
    __shared__ __align__(16) bf16 Bs[2][128 * 40];

    const int tid  = threadIdx.x;
    const int lane = tid & 31;
    const int wid  = tid >> 5;
    const int wm   = wid & 3;               // M wave offset: 32*wm
    const int wn   = wid >> 2;              // N wave offset: 64*wn
    const int ln   = lane & 15;
    const int hi   = lane >> 4;
    const int kb   = hi * 8;
    const int m0   = blockIdx.y * 128;
    const int n0   = blockIdx.x * 128;

    // staging coordinates: this thread owns chunks tid and tid+256
    // chunk c: row = c>>2 (0..127), col = (c&3)*8  (within the 32-wide slab)
    const int r0 = tid >> 2,         c0 = (tid & 3) * 8;
    const int r1 = (tid + 256) >> 2, c1 = ((tid + 256) & 3) * 8;

    const bf16* gA0 = A + (size_t)(m0 + r0) * K + c0;
    const bf16* gA1 = A + (size_t)(m0 + r1) * K + c1;
    const bf16* gB0 = B + (size_t)(n0 + r0) * K + c0;
    const bf16* gB1 = B + (size_t)(n0 + r1) * K + c1;

    uint32_t lA0[2], lA1[2], lB0[2], lB1[2];
#pragma unroll
    for (int bi = 0; bi < 2; ++bi) {
        lA0[bi] = lds_off32(&As[bi][r0 * LDT + c0]);
        lA1[bi] = lds_off32(&As[bi][r1 * LDT + c1]);
        lB0[bi] = lds_off32(&Bs[bi][r0 * LDT + c0]);
        lB1[bi] = lds_off32(&Bs[bi][r1 * LDT + c1]);
    }

    auto issue_stage = [&](int s) {
        const int buf = s & 1;
        const int k0  = s * 32;
        async_ld_b128(lA0[buf], gA0 + k0);
        async_ld_b128(lA1[buf], gA1 + k0);
        async_ld_b128(lB0[buf], gB0 + k0);
        async_ld_b128(lB1[buf], gB1 + k0);
    };

    v8f acc[2][4];
#pragma unroll
    for (int mt = 0; mt < 2; ++mt)
#pragma unroll
        for (int nt = 0; nt < 4; ++nt)
#pragma unroll
            for (int r = 0; r < 8; ++r) acc[mt][nt][r] = 0.0f;

    const int nk = K / 32;
    issue_stage(0);
    if (nk > 1) issue_stage(1);

    for (int k = 0; k < nk; ++k) {
        // stage k complete when at most the 4 ops of stage k+1 are outstanding
        if (k + 1 < nk) asm volatile("s_wait_asynccnt 0x4" ::: "memory");
        else            asm volatile("s_wait_asynccnt 0x0" ::: "memory");
        __syncthreads();

        const bf16* as = As[k & 1];
        const bf16* bs = Bs[k & 1];
        v16bf af[2], bfr[4];
#pragma unroll
        for (int mt = 0; mt < 2; ++mt)
            af[mt] = load_frag(&as[(wm * 32 + mt * 16 + ln) * LDT], kb);
#pragma unroll
        for (int nt = 0; nt < 4; ++nt)
            bfr[nt] = load_frag(&bs[(wn * 64 + nt * 16 + ln) * LDT], kb);

#pragma unroll
        for (int mt = 0; mt < 2; ++mt)
#pragma unroll
            for (int nt = 0; nt < 4; ++nt)
                acc[mt][nt] = WMMA_BF16(af[mt], bfr[nt], acc[mt][nt]);

        __syncthreads();                    // all waves done reading buf (k&1)
        if (k + 2 < nk) issue_stage(k + 2); // refill freed buffer
    }

    // epilogue
#pragma unroll
    for (int mt = 0; mt < 2; ++mt) {
#pragma unroll
        for (int nt = 0; nt < 4; ++nt) {
            const int gn = n0 + wn * 64 + nt * 16 + ln;
#pragma unroll
            for (int r = 0; r < 8; ++r) {
                const int gm = m0 + wm * 32 + mt * 16 + r + 8 * hi;
                float v = acc[mt][nt][r];
                if constexpr (HAS_BIAS) v += bias[gn];
                if constexpr (HAS_RELU) v = fmaxf(v, 0.0f);
                if constexpr (HAS_RES)  v += res[(size_t)gm * N + gn];
                if constexpr (OUT_BF16) Cb[(size_t)gm * N + gn] = (bf16)v;
                else                    Cf[(size_t)gm * N + gn] = v;
            }
        }
    }
}

// ---------------------------------------------------------------------------
// Flash attention (bf16 WMMA, f32 softmax state).
// grid: (SEQ/128, BATCH*NHEAD). 8 waves; each wave owns 16 query rows.
// KV tiles of 64 keys staged in LDS; V transposed at load.
// ---------------------------------------------------------------------------
__global__ __launch_bounds__(256) void flash_attn_wmma(
    const bf16* __restrict__ Q, const bf16* __restrict__ Km,
    const bf16* __restrict__ Vm, const int* __restrict__ mask,
    bf16* __restrict__ O)
{
    const int LDK = 72;                       // padded stride (halfs): 144B
    __shared__ __align__(16) bf16 Ks[64 * 72];
    __shared__ __align__(16) bf16 Vt[64 * 72];          // Vt[d][key]
    __shared__ __align__(16) bf16 Ps[8 * 16 * 72];      // per-wave P tile

    const int bh   = blockIdx.y;
    const int b    = bh >> 4;
    const int h    = bh & 15;
    const int tid  = threadIdx.x;
    const int lane = tid & 31;
    const int wid  = tid >> 5;
    const int ln   = lane & 15;
    const int hi   = lane >> 4;
    const int kb   = hi * 8;
    const int q0   = blockIdx.x * 128 + wid * 16;       // query offset in seq
    const size_t dbase = (size_t)h * DKH;

    // Q fragments (2 K-slabs of 32 over DK=64), straight from global
    v16bf qf[2];
    {
        const size_t qrow = (size_t)(b * SEQ + q0 + ln) * D_MODEL + dbase;
#pragma unroll
        for (int s = 0; s < 2; ++s)
            qf[s] = load_frag(Q + qrow + s * 32, kb);
    }

    v8f oacc[4];
#pragma unroll
    for (int n = 0; n < 4; ++n)
#pragma unroll
        for (int r = 0; r < 8; ++r) oacc[n][r] = 0.0f;

    float mrow[8], lrow[8];
#pragma unroll
    for (int r = 0; r < 8; ++r) { mrow[r] = -3.0e38f; lrow[r] = 0.0f; }

    const float scale = 0.125f;               // 1/sqrt(64)

    for (int kk0 = 0; kk0 < SEQ; kk0 += 64) {
        __syncthreads();
        // stage K tile [64][64] and V transposed [d][key]
#pragma unroll
        for (int c = tid; c < 512; c += 256) {
            const int r   = c >> 3;
            const int col = (c & 7) * 8;
            const size_t grow = (size_t)(b * SEQ + kk0 + r) * D_MODEL + dbase + col;
            *(uint4*)(&Ks[r * LDK + col]) = *(const uint4*)(Km + grow);
            v8bf vv = *(const v8bf*)(Vm + grow);
#pragma unroll
            for (int j = 0; j < 8; ++j) Vt[(col + j) * LDK + r] = vv[j];
        }
        __syncthreads();

        // scores S = Q * K^T  (4 tiles of 16 keys)
        v8f sacc[4];
#pragma unroll
        for (int t = 0; t < 4; ++t)
#pragma unroll
            for (int r = 0; r < 8; ++r) sacc[t][r] = 0.0f;
#pragma unroll
        for (int s = 0; s < 2; ++s)
#pragma unroll
            for (int t = 0; t < 4; ++t) {
                v16bf kf = load_frag(&Ks[(t * 16 + ln) * LDK + s * 32], kb);
                sacc[t] = WMMA_BF16(qf[s], kf, sacc[t]);
            }

        // scale + mask
        float sv[4][8];
#pragma unroll
        for (int t = 0; t < 4; ++t) {
            const int key = kk0 + t * 16 + ln;
            const int mk  = mask[b * SEQ + key];
#pragma unroll
            for (int r = 0; r < 8; ++r) {
                float xs = sacc[t][r] * scale;
                sv[t][r] = (mk == 0) ? -1.0e9f : xs;
            }
        }

        // online softmax: row stats live per lane-group of 16
        float mnew[8], alpha[8];
#pragma unroll
        for (int r = 0; r < 8; ++r) {
            float mx = fmaxf(fmaxf(sv[0][r], sv[1][r]), fmaxf(sv[2][r], sv[3][r]));
#pragma unroll
            for (int off = 8; off > 0; off >>= 1)
                mx = fmaxf(mx, __shfl_xor(mx, off, 32));
            mnew[r]  = fmaxf(mrow[r], mx);
            alpha[r] = __expf(mrow[r] - mnew[r]);
        }
        float ps[4][8];
#pragma unroll
        for (int t = 0; t < 4; ++t)
#pragma unroll
            for (int r = 0; r < 8; ++r)
                ps[t][r] = __expf(sv[t][r] - mnew[r]);
#pragma unroll
        for (int r = 0; r < 8; ++r) {
            float sm = ps[0][r] + ps[1][r] + ps[2][r] + ps[3][r];
#pragma unroll
            for (int off = 8; off > 0; off >>= 1)
                sm += __shfl_xor(sm, off, 32);
            lrow[r] = lrow[r] * alpha[r] + sm;
            mrow[r] = mnew[r];
        }
#pragma unroll
        for (int n = 0; n < 4; ++n)
#pragma unroll
            for (int r = 0; r < 8; ++r) oacc[n][r] *= alpha[r];

        // restage P as bf16 A-fragments (per-wave LDS region, same-wave in-order)
        bf16* pw = &Ps[wid * 16 * LDK];
#pragma unroll
        for (int t = 0; t < 4; ++t)
#pragma unroll
            for (int r = 0; r < 8; ++r)
                pw[(r + 8 * hi) * LDK + t * 16 + ln] = (bf16)ps[t][r];

        // O += P * V
#pragma unroll
        for (int s = 0; s < 2; ++s) {
            v16bf pf = load_frag(&pw[ln * LDK + s * 32], kb);
#pragma unroll
            for (int n = 0; n < 4; ++n) {
                v16bf vf = load_frag(&Vt[(n * 16 + ln) * LDK + s * 32], kb);
                oacc[n] = WMMA_BF16(pf, vf, oacc[n]);
            }
        }
    }

    // finalize: O /= l, write bf16 [token][h*64 + d]
#pragma unroll
    for (int r = 0; r < 8; ++r) {
        const float invl = 1.0f / lrow[r];
        const size_t gm  = (size_t)(b * SEQ + q0 + r + 8 * hi);
#pragma unroll
        for (int n = 0; n < 4; ++n)
            O[gm * D_MODEL + dbase + n * 16 + ln] = (bf16)(oacc[n][r] * invl);
    }
}

// ---------------------------------------------------------------------------
// launch
// ---------------------------------------------------------------------------
extern "C" void kernel_launch(void* const* d_in, const int* in_sizes, int n_in,
                              void* d_out, int out_size, void* d_ws, size_t ws_size,
                              hipStream_t stream)
{
    const float* x      = (const float*)d_in[0];
    const int*   mask   = (const int*)  d_in[1];
    const float* w_q    = (const float*)d_in[2];
    const float* w_k    = (const float*)d_in[3];
    const float* w_v    = (const float*)d_in[4];
    const float* w_o    = (const float*)d_in[5];
    const float* alpha1 = (const float*)d_in[6];
    const float* beta1  = (const float*)d_in[7];
    const float* alpha2 = (const float*)d_in[8];
    const float* beta2  = (const float*)d_in[9];
    const float* fc1_w  = (const float*)d_in[10];
    const float* fc1_b  = (const float*)d_in[11];
    const float* fc2_w  = (const float*)d_in[12];
    const float* fc2_b  = (const float*)d_in[13];

    char* ws = (char*)d_ws;
    size_t off = 0;
    auto take = [&](size_t bytes) { char* p = ws + off; off += bytes; return p; };

    bf16* wqb   = (bf16*)take((size_t)D_MODEL * D_MODEL * 2);
    bf16* wkb   = (bf16*)take((size_t)D_MODEL * D_MODEL * 2);
    bf16* wvb   = (bf16*)take((size_t)D_MODEL * D_MODEL * 2);
    bf16* wob   = (bf16*)take((size_t)D_MODEL * D_MODEL * 2);
    bf16* fc1b  = (bf16*)take((size_t)D_FC * D_MODEL * 2);
    bf16* fc2b  = (bf16*)take((size_t)D_MODEL * D_FC * 2);
    bf16* ln1b  = (bf16*)take((size_t)M_TOK * D_MODEL * 2);
    bf16* Qb    = (bf16*)take((size_t)M_TOK * D_MODEL * 2);
    bf16* Kb    = (bf16*)take((size_t)M_TOK * D_MODEL * 2);
    bf16* Vb    = (bf16*)take((size_t)M_TOK * D_MODEL * 2);
    bf16* attnb = (bf16*)take((size_t)M_TOK * D_MODEL * 2);
    float* x1   = (float*)take((size_t)M_TOK * D_MODEL * 4);
    bf16* ln2b  = (bf16*)take((size_t)M_TOK * D_MODEL * 2);
    bf16* hidb  = (bf16*)take((size_t)M_TOK * D_FC * 2);

    // weight conversions
    {
        int n4 = D_MODEL * D_MODEL / 4;
        cvt_f32_bf16<<<(n4 + 255) / 256, 256, 0, stream>>>(w_q, wqb, n4);
        cvt_f32_bf16<<<(n4 + 255) / 256, 256, 0, stream>>>(w_k, wkb, n4);
        cvt_f32_bf16<<<(n4 + 255) / 256, 256, 0, stream>>>(w_v, wvb, n4);
        cvt_f32_bf16<<<(n4 + 255) / 256, 256, 0, stream>>>(w_o, wob, n4);
        int nf = D_FC * D_MODEL / 4;
        cvt_f32_bf16<<<(nf + 255) / 256, 256, 0, stream>>>(fc1_w, fc1b, nf);
        cvt_f32_bf16<<<(nf + 255) / 256, 256, 0, stream>>>(fc2_w, fc2b, nf);
    }

    // LN1
    layernorm_bf16<<<M_TOK, 256, 0, stream>>>(x, alpha1, beta1, ln1b);

    // Q, K, V projections  (bf16 out)
    dim3 gP(D_MODEL / 128, M_TOK / 128);
    gemm_bf16_wmma<0,0,0,1><<<gP, 256, 0, stream>>>(ln1b, wqb, nullptr, nullptr,
                                                    nullptr, Qb, M_TOK, D_MODEL, D_MODEL);
    gemm_bf16_wmma<0,0,0,1><<<gP, 256, 0, stream>>>(ln1b, wkb, nullptr, nullptr,
                                                    nullptr, Kb, M_TOK, D_MODEL, D_MODEL);
    gemm_bf16_wmma<0,0,0,1><<<gP, 256, 0, stream>>>(ln1b, wvb, nullptr, nullptr,
                                                    nullptr, Vb, M_TOK, D_MODEL, D_MODEL);

    // attention
    flash_attn_wmma<<<dim3(SEQ / 128, BATCH * NHEAD), 256, 0, stream>>>(
        Qb, Kb, Vb, mask, attnb);

    // output projection + residual:  x1 = x + attn @ w_o^T  (f32 out)
    gemm_bf16_wmma<0,0,1,0><<<gP, 256, 0, stream>>>(attnb, wob, nullptr, x,
                                                    x1, nullptr, M_TOK, D_MODEL, D_MODEL);

    // LN2
    layernorm_bf16<<<M_TOK, 256, 0, stream>>>(x1, alpha2, beta2, ln2b);

    // FC1: relu(ln2 @ fc1^T + b1) -> bf16 hidden
    dim3 gF1(D_FC / 128, M_TOK / 128);
    gemm_bf16_wmma<1,1,0,1><<<gF1, 256, 0, stream>>>(ln2b, fc1b, fc1_b, nullptr,
                                                     nullptr, hidb, M_TOK, D_FC, D_MODEL);

    // FC2: out = x1 + hidden @ fc2^T + b2  (f32 -> d_out)
    gemm_bf16_wmma<1,0,1,0><<<gP, 256, 0, stream>>>(hidb, fc2b, fc2_b, x1,
                                                    (float*)d_out, nullptr, M_TOK, D_MODEL, D_FC);

    (void)in_sizes; (void)n_in; (void)out_size; (void)ws_size;
}
